// SparseModel_89618787598436
// MI455X (gfx1250) — matrix operand order repaired and checked by reading, not compiled, not run on hardware
//
#include <hip/hip_runtime.h>
#include <hip/hip_bf16.h>

// ---------------------------------------------------------------------------
// SparseModel on MI455X (gfx1250).
//   stage:  out[M,N] = A4[M,4S] @ W[4S,N]
//     A4[b, 4*s+t] = f_t(w * src[b,s])          (bf16, built on the fly)
//     W [4*s+t, n] = (mat[n,s] == t+1)          (bf16 exact 0/1)
// Interleaved K (4*s+t): one x-tile load feeds all 4 activation slabs.
// ~190 GFLOP vs ~0.2 GB traffic -> compute-bound -> bf16 WMMA, f32 accum.
// tanh/sigmoid use gfx1250 V_TANH_F32 when available.
// B (mask) tiles stream global->LDS via GLOBAL_LOAD_ASYNC_TO_LDS_B128
// (ASYNCcnt-tracked, no VGPR staging, no ds_store issue traffic).
// ---------------------------------------------------------------------------

typedef __attribute__((ext_vector_type(16))) __bf16 v16bf;
typedef __attribute__((ext_vector_type(8)))  __bf16 v8bf;
typedef __attribute__((ext_vector_type(8)))  float  v8f;

#define BATCH   8192
#define IN_DIM  1024
#define N_HID   2048
#define OUT_DIM 256
#define SRC_DIM (IN_DIM + N_HID)   // 3072

#define USE_ASYNC_B 1              // 0 -> VGPR-staged B path (fallback)

#if defined(__has_builtin)
# if __has_builtin(__builtin_amdgcn_tanhf)
#  define TANHF(x) __builtin_amdgcn_tanhf(x)
# elif __has_builtin(__builtin_amdgcn_tanh_f32)
#  define TANHF(x) __builtin_amdgcn_tanh_f32(x)
# endif
#endif
#ifndef TANHF
# define TANHF(x) tanhf(x)
#endif

__device__ __forceinline__ unsigned int f2bf(float f) {
    unsigned u = __float_as_uint(f);
    u += 0x7FFFu + ((u >> 16) & 1u);          // round-to-nearest-even
    return u >> 16;
}

// All 4 activations of one value, packed as 4 bf16 (codes 1..4 -> t 0..3).
__device__ __forceinline__ uint2 act4(float z) {
    unsigned a0 = f2bf(z);
    unsigned a1 = f2bf(fmaxf(z, 0.0f));
    unsigned a2 = f2bf(TANHF(z));
    unsigned a3 = f2bf(0.5f + 0.5f * TANHF(0.5f * z));   // sigmoid
    uint2 d;
    d.x = a0 | (a1 << 16);
    d.y = a2 | (a3 << 16);
    return d;
}

// Generic pointer into __shared__ -> 32-bit LDS offset (ISA 10.2 aperture:
// low 32 bits of the shared-aperture generic address are the LDS address).
__device__ __forceinline__ unsigned lds_off(const void* p) {
    return (unsigned)(unsigned long long)p;
}

// 16B global -> 16B LDS, per lane, ASYNCcnt-tracked.
__device__ __forceinline__ void async_b128(unsigned lds, unsigned goff,
                                           unsigned long long gbase) {
#if USE_ASYNC_B
    asm volatile("global_load_async_to_lds_b128 %0, %1, %2"
                 :: "v"(lds), "v"(goff), "s"(gbase)
                 : "memory");
#endif
}

__device__ __forceinline__ void wait_async0() {
#if USE_ASYNC_B
    asm volatile("s_wait_asynccnt 0" ::: "memory");
#endif
}

// ---------------------------------------------------------------------------
// Mask-weight build (interleaved K): W[(4*s+t)*N + n] = (mat[n,s] == t+1).
// All index math is power-of-2. Writes coalesced; mat reads L2-resident.
// ---------------------------------------------------------------------------
__global__ void build_masks(const int* __restrict__ mat,
                            unsigned short* __restrict__ W1,
                            unsigned short* __restrict__ W2) {
    const long long T1 = 4LL * IN_DIM * N_HID;
    const long long T2 = 4LL * SRC_DIM * OUT_DIM;
    long long idx = (long long)blockIdx.x * blockDim.x + threadIdx.x;
    if (idx < T1) {
        int n  = (int)(idx & (N_HID - 1));
        int k4 = (int)(idx >> 11);            // / N_HID
        int tt = k4 & 3, s = k4 >> 2;
        int code = mat[(size_t)n * SRC_DIM + s];
        W1[idx] = (code == tt + 1) ? (unsigned short)0x3F80 : (unsigned short)0;
    } else if (idx < T1 + T2) {
        long long j = idx - T1;
        int n  = (int)(j & (OUT_DIM - 1));
        int k4 = (int)(j >> 8);               // / OUT_DIM
        int tt = k4 & 3, s = k4 >> 2;
        int code = mat[(size_t)(N_HID + n) * SRC_DIM + s];
        W2[j] = (code == tt + 1) ? (unsigned short)0x3F80 : (unsigned short)0;
    }
}

// ---------------------------------------------------------------------------
// Activation-expanded bf16 WMMA GEMM, K4 = 128 per iteration (4 WMMA k-steps).
// Block tile BM x 128; wave tile 64x32 = 4x2 16x16 accumulators.
// ---------------------------------------------------------------------------
template <int BM, int N, int S, int SPLIT, int LDA, int LDB>
__global__ __launch_bounds__(32 * (BM / 64) * 4)
void act_wmma_gemm(const float* __restrict__ srcA,
                   const float* __restrict__ srcB,
                   const unsigned short* __restrict__ Wm,
                   const float* __restrict__ wscale,
                   float* __restrict__ out) {
    constexpr int BN  = 128;
    constexpr int BKS = 32;                   // source columns per iteration
    constexpr int BK4 = 4 * BKS;              // expanded K per iteration = 128
    constexpr int NT  = 32 * (BM / 64) * 4;   // threads per block
    constexpr int AST = BK4 + 8;              // LDS row strides (+16B pad:
    constexpr int BST = BN + 8;               //  conflict-free b128 reads)
    constexpr int BPT = (BK4 * BN) / NT;      // B bf16 per thread (64 / 128)
    constexpr int BCH = BPT / 8;              // 16-byte chunks per thread

    __shared__ __align__(16) unsigned short As[2][BM][AST];
    __shared__ __align__(16) unsigned short Bs[2][BK4][BST];

    const int tid  = threadIdx.x;
    const int lane = tid & 31;
    const int wid  = tid >> 5;
    const int wm   = wid % (BM / 64);
    const int wn   = wid / (BM / 64);
    const int bx   = blockIdx.x, by = blockIdx.y;
    const float wv = *wscale;

    v8f acc[4][2];
    #pragma unroll
    for (int i = 0; i < 4; ++i)
        #pragma unroll
        for (int j = 0; j < 2; ++j)
            #pragma unroll
            for (int k = 0; k < 8; ++k) acc[i][j][k] = 0.0f;

    // Loader coordinates: A = BM x 32 f32 tile, 16 floats/thread.
    const int ar = tid >> 1;
    const int ac = (tid & 1) * 16;
    // B = 128 x BN bf16 tile, BPT contiguous bf16/thread.
    const int br = (tid * BPT) / BN;
    const int bc = (tid * BPT) % BN;

    constexpr int nk = S / BKS;

    const unsigned long long wbase = (unsigned long long)Wm;

    float4 aR[4];
#if !USE_ASYNC_B
    uint4  bR[BCH];
#endif

    // Global A loads into registers (activation applied at stage time).
    auto fetchA = [&](int kb) {
        const int sg0 = kb * BKS;             // column in concat [x | hidden]
        const float* p;
        if (sg0 < SPLIT) p = srcA + (size_t)(by * BM + ar) * LDA + sg0 + ac;
        else             p = srcB + (size_t)(by * BM + ar) * LDB + (sg0 - SPLIT) + ac;
        #pragma unroll
        for (int i = 0; i < 4; ++i) aR[i] = ((const float4*)p)[i];
    };

    // B tile: global -> LDS, bypassing VGPRs (ASYNCcnt) or staged (fallback).
    auto fetchB = [&](int kb, int buf) {
        const unsigned gbyte =
            (unsigned)(((kb * BK4 + br) * N + bx * BN + bc) * 2);
#if USE_ASYNC_B
        const unsigned lbyte = lds_off(&Bs[buf][br][bc]);
        #pragma unroll
        for (int c = 0; c < BCH; ++c)
            async_b128(lbyte + 16u * c, gbyte + 16u * c, wbase);
#else
        const unsigned short* q = Wm + gbyte / 2;
        #pragma unroll
        for (int c = 0; c < BCH; ++c) bR[c] = ((const uint4*)q)[c];
#endif
    };

    auto stage = [&](int buf) {
        // Each source float expands to 4 consecutive bf16 (t = 0..3).
        unsigned short* arow = &As[buf][ar][0];
        #pragma unroll
        for (int i = 0; i < 4; ++i) {
            float4 v = aR[i];
            *(uint2*)(arow + 4 * (ac + 4 * i + 0)) = act4(wv * v.x);
            *(uint2*)(arow + 4 * (ac + 4 * i + 1)) = act4(wv * v.y);
            *(uint2*)(arow + 4 * (ac + 4 * i + 2)) = act4(wv * v.z);
            *(uint2*)(arow + 4 * (ac + 4 * i + 3)) = act4(wv * v.w);
        }
#if !USE_ASYNC_B
        #pragma unroll
        for (int c = 0; c < BCH; ++c)
            *(uint4*)&Bs[buf][br][bc + 8 * c] = bR[c];
#endif
    };

    auto compute = [&](int buf) {
        const unsigned short* Ab = &As[buf][0][0];
        const unsigned short* Bb = &Bs[buf][0][0];
        const int half = lane >> 4;
        const int ml   = lane & 15;
        #pragma unroll
        for (int kk = 0; kk < 4; ++kk) {      // 4 WMMA k-steps per buffer
            v16bf af[4], bfr[2];
            // A 16x32 bf16 frag (ISA 7.12.2): lane m = lane&15;
            // VGPR0-3: K = h*8+0..7, VGPR4-7: K = 16+h*8+0..7
            #pragma unroll
            for (int i = 0; i < 4; ++i) {
                const int row = wm * 64 + i * 16 + ml;
                const unsigned short* p0 = Ab + row * AST + kk * 32 + half * 8;
                const v8bf lo = *(const v8bf*)p0;
                const v8bf hi = *(const v8bf*)(p0 + 16);
                #pragma unroll
                for (int k = 0; k < 8; ++k) { af[i][k] = lo[k]; af[i][k + 8] = hi[k]; }
            }
            // B 32x16 bf16 frag: lane n = lane&15; K = h*8+0..7, 16+h*8+0..7
            #pragma unroll
            for (int j = 0; j < 2; ++j) {
                const int col = wn * 32 + j * 16 + ml;
                const unsigned short* p = Bb + (kk * 32 + half * 8) * BST + col;
                #pragma unroll
                for (int k = 0; k < 8; ++k) {
                    bfr[j][k]     = __builtin_bit_cast(__bf16, p[k * BST]);
                    bfr[j][k + 8] = __builtin_bit_cast(__bf16, p[(k + 16) * BST]);
                }
            }
            #pragma unroll
            for (int i = 0; i < 4; ++i)
                #pragma unroll
                for (int j = 0; j < 2; ++j)
                    acc[i][j] = __builtin_amdgcn_wmma_f32_16x16x32_bf16(
                        false, af[i], false, bfr[j], (short)0, acc[i][j],
                        false, false);
        }
    };

    // Prologue: fill buffer 0.
    fetchA(0);
    fetchB(0, 0);
    stage(0);
    wait_async0();
    __syncthreads();

    for (int kb = 0; kb < nk; ++kb) {
        const int cur = kb & 1;
        if (kb + 1 < nk) {
            fetchA(kb + 1);
            fetchB(kb + 1, cur ^ 1);          // async: lands in LDS directly
        }
        if (kb + 2 < nk)                      // stream hint, mask tile after next
            __builtin_prefetch(Wm + (size_t)((kb + 2) * BK4) * N + bx * BN, 0, 0);
        compute(cur);
        if (kb + 1 < nk) stage(cur ^ 1);
        wait_async0();                        // B[nxt] resident before barrier
        __syncthreads();
    }

    // C/D 16x16 f32 (ISA 7.12.2): VGPR v -> M = v + 8*(lane>=16); N = lane&15
    #pragma unroll
    for (int i = 0; i < 4; ++i) {
        #pragma unroll
        for (int j = 0; j < 2; ++j) {
            const int m0 = by * BM + wm * 64 + i * 16 + (lane >> 4) * 8;
            const int n0 = bx * BN + wn * 32 + j * 16 + (lane & 15);
            #pragma unroll
            for (int v = 0; v < 8; ++v)
                out[(size_t)(m0 + v) * N + n0] = acc[i][j][v];
        }
    }
}

// ---------------------------------------------------------------------------
// Workspace layout (86 MB total):
//   [0,     16 MB)  W1 bf16 [4096 x 2048]   (K interleaved 4*s+t)
//   [16 MB, 22 MB)  W2 bf16 [12288 x 256]
//   [22 MB, 86 MB)  hidden f32 [8192 x 2048]
// ---------------------------------------------------------------------------
extern "C" void kernel_launch(void* const* d_in, const int* in_sizes, int n_in,
                              void* d_out, int out_size, void* d_ws, size_t ws_size,
                              hipStream_t stream) {
    (void)in_sizes; (void)n_in; (void)out_size; (void)ws_size;
    const float* x   = (const float*)d_in[0];
    const float* w   = (const float*)d_in[1];
    const int*   mat = (const int*)d_in[2];
    float*       out = (float*)d_out;

    char* ws = (char*)d_ws;
    const size_t w1_bytes = (size_t)4 * IN_DIM * N_HID * sizeof(unsigned short);
    const size_t w2_bytes = (size_t)4 * SRC_DIM * OUT_DIM * sizeof(unsigned short);
    unsigned short* W1 = (unsigned short*)ws;
    unsigned short* W2 = (unsigned short*)(ws + w1_bytes);
    float* hidden      = (float*)(ws + w1_bytes + w2_bytes);

    const long long total = 4LL * IN_DIM * N_HID + 4LL * SRC_DIM * OUT_DIM;
    build_masks<<<(int)((total + 255) / 256), 256, 0, stream>>>(mat, W1, W2);

    // Stage 1: hidden[8192,2048] = A4(x)[8192,4096] @ W1
    act_wmma_gemm<128, N_HID, IN_DIM, IN_DIM, IN_DIM, IN_DIM>
        <<<dim3(N_HID / 128, BATCH / 128), 256, 0, stream>>>(
            x, x, W1, w, hidden);

    // Stage 2: out[8192,256] = A4([x|hidden])[8192,12288] @ W2
    // BM=64 -> 256 blocks to keep the machine full despite N=256.
    act_wmma_gemm<64, OUT_DIM, SRC_DIM, IN_DIM, IN_DIM, N_HID>
        <<<dim3(OUT_DIM / 128, BATCH / 64), 128, 0, stream>>>(
            x, hidden, W2, w, out);
}
